// ModulatedConv2d_11879879543714
// MI455X (gfx1250) — matrix operand order repaired and checked by reading, not compiled, not run on hardware
//
#include <hip/hip_runtime.h>

// Problem constants
#define CIN   512
#define COUT  512
#define HID   512
#define BATCH 16
#define HH    64
#define WW    64
#define KCH   32                  // cin per chunk (WMMA K)
#define NCHUNK (CIN / KCH)        // 16
#define APAD  40                  // padded K-dim in LDS (80B rows: 16B aligned, conflict-free)

typedef __attribute__((ext_vector_type(16))) __bf16 v16bf;
typedef __attribute__((ext_vector_type(8)))  float  v8f;
typedef __attribute__((ext_vector_type(4)))  unsigned int u32x4;
typedef __attribute__((ext_vector_type(4)))  int i32x4;
typedef __attribute__((ext_vector_type(8)))  int i32x8;

struct __align__(16) U128 { unsigned int x, y, z, w; };
union  FragBF { U128 u[2]; v16bf v; };
struct __align__(16) F4  { float x, y, z, w; };
struct __align__(8)  US4 { unsigned short x, y, z, w; };

#if __has_builtin(__builtin_amdgcn_tensor_load_to_lds) && __has_builtin(__builtin_amdgcn_s_wait_tensorcnt)
#define USE_TDM 1
#else
#define USE_TDM 0
#endif

__device__ __forceinline__ unsigned short f2bf(float f) {
  unsigned int u = __float_as_uint(f);
  u += 0x7FFFu + ((u >> 16) & 1u);
  return (unsigned short)(u >> 16);
}

// LDS aperture is 4GB-aligned,
// so the low 32 bits of a flat pointer into __shared__ are the LDS byte offset.
__device__ __forceinline__ unsigned lds_byte_off(const void* p) {
  return (unsigned)(unsigned long long)p;
}

#if USE_TDM
// TDM: load 9 taps x 4KB (64 cout x 32 cin bf16) at 512KB stride into LDS,
// hardware-padding 4 DWORDs after every 16 DWORDs to produce the APAD=40 layout.
__device__ __forceinline__ void tdm_load_A(const unsigned short* gsrc, unsigned ldsoff) {
  unsigned long long ga = (unsigned long long)gsrc;
  u32x4 g0;
  g0[0] = 1u;                                            // count=1 (valid D#)
  g0[1] = ldsoff;                                        // lds_addr (bytes)
  g0[2] = (unsigned)ga;                                  // global_addr[31:0]
  g0[3] = (unsigned)((ga >> 32) & 0x1FFFFFFu) | (2u << 30); // global_addr[56:32], type=2
  i32x8 g1;
  g1[0] = (int)((3u << 16)    // data_size = 8B units
              | (1u << 20)    // pad_enable
              | (3u << 22)    // pad_interval: 16 DWORDs (64B)
              | (3u << 25));  // pad_amount: 4 DWORDs (16B)
  g1[1] = (int)(512u << 16);  // tensor_dim0[15:0]=512 (in 8B units) in high half
  g1[2] = (int)(9u << 16);    // tensor_dim0 hi=0 | tensor_dim1[15:0]=9
  g1[3] = (int)(512u << 16);  // tensor_dim1 hi=0 | tile_dim0=512 (4KB row)
  g1[4] = 9;                  // tile_dim1=9 | tile_dim2=0
  g1[5] = 65536;              // tensor_dim0_stride lo32 = 512KB/8B
  g1[6] = 0;
  g1[7] = 0;
  i32x4 z4 = {0, 0, 0, 0};    // groups 2/3 unused (2D tensor)
  i32x8 z8 = {0, 0, 0, 0, 0, 0, 0, 0};
  __builtin_amdgcn_tensor_load_to_lds(g0, g1, z4, z4, z8, 0);
}
#endif

// -------- kernel 1: style s[b,i] = (y @ mod_w.T)/sqrt(HID) + mod_b + 1 --------
__global__ void style_kernel(const float* __restrict__ y,
                             const float* __restrict__ mod_w,
                             const float* __restrict__ mod_b,
                             float* __restrict__ s_out) {
  int idx = blockIdx.x * blockDim.x + threadIdx.x;
  if (idx >= BATCH * CIN) return;
  int b = idx >> 9, i = idx & (CIN - 1);
  const float* yr = y + b * HID;
  const float* wr = mod_w + i * HID;
  float acc = 0.f;
  #pragma unroll 4
  for (int h = 0; h < HID; ++h) acc = fmaf(yr[h], wr[h], acc);
  s_out[idx] = acc * 0.044194173824159216f + mod_b[i] + 1.0f;
}

// -------- kernel 2: w2[o,i] = sum_t (w*coef)^2 ; pack bf16 weights --------
// wpack layout: [tap 9][chunk 16][cout 512][32]
__global__ void w2pack_kernel(const float* __restrict__ weight,
                              float* __restrict__ w2,
                              unsigned short* __restrict__ wpack) {
  int idx = blockIdx.x * blockDim.x + threadIdx.x;
  if (idx >= COUT * CIN) return;
  int o = idx >> 9, i = idx & (CIN - 1);
  const float coef = 0.014731391274719739f;   // 1/sqrt(512*9)
  int chunk = i >> 5, il = i & 31;
  const float* wr = weight + (size_t)(o * CIN + i) * 9;
  float acc = 0.f;
  #pragma unroll
  for (int t = 0; t < 9; ++t) {
    float v = wr[t] * coef;
    acc = fmaf(v, v, acc);
    wpack[(((size_t)(t * NCHUNK + chunk) * COUT + o) << 5) + il] = f2bf(v);
  }
  w2[idx] = acc;
}

// -------- kernel 3: d[b,o] = rsqrt(sum_i w2[o,i]*s[b,i]^2 + 1e-8) --------
__global__ void demod_kernel(const float* __restrict__ w2,
                             const float* __restrict__ s_in,
                             float* __restrict__ d_out) {
  int idx = blockIdx.x * blockDim.x + threadIdx.x;
  if (idx >= BATCH * COUT) return;
  int b = idx >> 9, o = idx & (COUT - 1);
  const float* w2r = w2 + o * CIN;
  const float* sr  = s_in + b * CIN;
  float acc = 0.f;
  #pragma unroll 4
  for (int i = 0; i < CIN; ++i) { float sv = sr[i]; acc = fmaf(w2r[i], sv * sv, acc); }
  d_out[idx] = rsqrtf(acc + 1e-8f);
}

// -------- kernel 4: xmb[b,i,h,w] = bf16(x * s[b,i]) --------
__global__ void xmpack_kernel(const float* __restrict__ x,
                              const float* __restrict__ s_in,
                              unsigned short* __restrict__ xmb) {
  long long idx = (long long)blockIdx.x * blockDim.x + threadIdx.x;
  long long e = idx * 4;
  if (e >= (long long)BATCH * CIN * HH * WW) return;
  int b = (int)(e >> 21);
  int i = (int)(e >> 12) & (CIN - 1);
  float sc = s_in[b * CIN + i];
  F4 xv = *(const F4*)(x + e);
  US4 o;
  o.x = f2bf(xv.x * sc); o.y = f2bf(xv.y * sc);
  o.z = f2bf(xv.z * sc); o.w = f2bf(xv.w * sc);
  *(US4*)(xmb + e) = o;
}

// -------- kernel 5: implicit-GEMM conv via WMMA bf16 + TDM staging --------
// grid: (COUT/64, HH/2, B), block 256 (8 waves).
// Block computes out[b, cout0..+63, h0..h0+1, 0..63]  (64 cout x 128 px).
__global__ __launch_bounds__(256) void conv_wmma_kernel(
    const unsigned short* __restrict__ wpack,
    const unsigned short* __restrict__ xmb,
    const float* __restrict__ dcoef,
    const float* __restrict__ noise,
    const float* __restrict__ bias,
    const float* __restrict__ noise_scale,
    float* __restrict__ out) {
  __shared__ __align__(16) unsigned short sA[2][9][64][APAD];  // double-buffered [tap][cout][k]
  __shared__ __align__(16) unsigned short sB[4][66][APAD];     // [window row h0-1..h0+2][col][cin]

  const int tid   = threadIdx.x;
  const int lane  = tid & 31;
  const int wave  = tid >> 5;
  const int l15   = lane & 15;
  const int hi    = lane >> 4;
  const int mt    = wave & 3;                // cout tile 0..3
  const int nb    = (wave >> 2) * 2;         // column tiles nb, nb+1
  const int cout0 = blockIdx.x * 64;
  const int h0    = blockIdx.y * 2;
  const int b     = blockIdx.z;

  // zero horizontal halo columns once
  for (int i = tid; i < 4 * KCH; i += 256) {
    int wr = i >> 5, ci = i & 31;
    sB[wr][0][ci] = 0; sB[wr][65][ci] = 0;
  }

  v8f acc[2][2] = {};   // [output row r][col tile j]

#if USE_TDM
  if (wave == 0)
    tdm_load_A(wpack + ((size_t)(0 * 512 + cout0) << 5), lds_byte_off(&sA[0][0][0][0]));
#endif

  for (int chunk = 0; chunk < NCHUNK; ++chunk) {
    const int cur = chunk & 1;
    __syncthreads();   // previous compute done: safe to overwrite sB (and sA in fallback)

#if !USE_TDM
    // synchronous A staging fallback
    for (int i = tid; i < 9 * 1024; i += 256) {
      int t = i >> 10, r = i & 1023;
      int m = r >> 4, kp = r & 15;
      const unsigned int* src = (const unsigned int*)wpack +
          ((size_t)((t * NCHUNK + chunk) * COUT + cout0) << 4);
      ((unsigned int*)&sA[cur][t][m][0])[kp] = src[r];
    }
#endif
    // stage B: 4 window rows x 32 cin x 64 px, transposed to [col][cin]
    const int cin0 = chunk * KCH;
    for (int i = tid; i < 4 * 1024; i += 256) {
      int wr = i >> 10, r = i & 1023;
      int ci = r >> 5, pp = r & 31;
      int px = pp * 2;
      int h_in = h0 + wr - 1;
      unsigned int val = 0;
      if (h_in >= 0 && h_in < HH)
        val = *(const unsigned int*)(xmb +
              ((size_t)((b * CIN + cin0 + ci) * HH + h_in) << 6) + px);
      sB[wr][px + 1][ci] = (unsigned short)(val & 0xFFFFu);
      sB[wr][px + 2][ci] = (unsigned short)(val >> 16);
    }
#if USE_TDM
    if (wave == 0) __builtin_amdgcn_s_wait_tensorcnt(0);   // sA[cur] DMA complete
#endif
    __syncthreads();
#if USE_TDM
    if (wave == 0 && chunk + 1 < NCHUNK)                   // prefetch next chunk's A
      tdm_load_A(wpack + ((size_t)((chunk + 1) * 512 + cout0) << 5),
                 lds_byte_off(&sA[cur ^ 1][0][0][0]));
#endif

    #pragma unroll
    for (int t = 0; t < 9; ++t) {
      const int kh = t / 3, kw = t % 3;
      FragBF a;
      const U128* ap = (const U128*)&sA[cur][t][mt * 16 + l15][0];
      a.u[0] = ap[hi];
      a.u[1] = ap[hi + 2];
      #pragma unroll
      for (int r = 0; r < 2; ++r) {
        #pragma unroll
        for (int j = 0; j < 2; ++j) {
          FragBF bf;
          const U128* bp = (const U128*)&sB[r + kh][(nb + j) * 16 + l15 + kw][0];
          bf.u[0] = bp[hi];
          bf.u[1] = bp[hi + 2];
          acc[r][j] = __builtin_amdgcn_wmma_f32_16x16x32_bf16(
              false, a.v, false, bf.v, (short)0, acc[r][j], false, false);
        }
      }
    }
  }

  // ---- epilogue: out = acc*d + noise*ns + bias ----
  const float ns = noise_scale[0];
  #pragma unroll
  for (int v = 0; v < 8; ++v) {
    int co = cout0 + mt * 16 + v + 8 * hi;   // D layout: M = v + 8*hi, N = lane&15
    float dd = dcoef[b * COUT + co];
    float bb = bias[co];
    #pragma unroll
    for (int r = 0; r < 2; ++r) {
      int h = h0 + r;
      #pragma unroll
      for (int j = 0; j < 2; ++j) {
        int px = (nb + j) * 16 + l15;
        float nz = noise[((b * HH + h) << 6) + px] * ns;
        out[(((size_t)(b * COUT + co) * HH + h) << 6) + px] = acc[r][j][v] * dd + nz + bb;
      }
    }
  }
}

extern "C" void kernel_launch(void* const* d_in, const int* in_sizes, int n_in,
                              void* d_out, int out_size, void* d_ws, size_t ws_size,
                              hipStream_t stream) {
  const float* x           = (const float*)d_in[0];
  const float* y           = (const float*)d_in[1];
  const float* noise       = (const float*)d_in[2];
  const float* weight      = (const float*)d_in[3];
  const float* bias        = (const float*)d_in[4];
  const float* mod_w       = (const float*)d_in[5];
  const float* mod_b       = (const float*)d_in[6];
  const float* noise_scale = (const float*)d_in[7];
  float* out = (float*)d_out;

  float* s_buf  = (float*)d_ws;                         // 8192
  float* d_buf  = s_buf + BATCH * CIN;                  // 8192
  float* w2_buf = d_buf + BATCH * COUT;                 // 262144
  unsigned short* wpack = (unsigned short*)(w2_buf + COUT * CIN);   // 9*512*512
  unsigned short* xmb   = wpack + (size_t)9 * COUT * CIN;           // 16*512*4096

  style_kernel<<<(BATCH * CIN + 255) / 256, 256, 0, stream>>>(y, mod_w, mod_b, s_buf);
  w2pack_kernel<<<(COUT * CIN + 255) / 256, 256, 0, stream>>>(weight, w2_buf, wpack);
  demod_kernel<<<(BATCH * COUT + 255) / 256, 256, 0, stream>>>(w2_buf, s_buf, d_buf);
  long long nx = (long long)BATCH * CIN * HH * WW / 4;
  xmpack_kernel<<<(unsigned)((nx + 255) / 256), 256, 0, stream>>>(x, s_buf, xmb);
  dim3 grid(COUT / 64, HH / 2, BATCH);
  conv_wmma_kernel<<<grid, 256, 0, stream>>>(wpack, xmb, d_buf, noise, bias,
                                             noise_scale, out);
}
